// Transformer_37323265802366
// MI455X (gfx1250) — matrix operand order repaired and checked by reading, not compiled, not run on hardware
//
#include <hip/hip_runtime.h>
#include <hip/hip_bf16.h>
#include <math.h>

typedef __bf16 bf16_t;
typedef bf16_t v16bf __attribute__((ext_vector_type(16)));
typedef bf16_t v8bf  __attribute__((ext_vector_type(8)));
typedef bf16_t v4bf  __attribute__((ext_vector_type(4)));
typedef float  v8f   __attribute__((ext_vector_type(8)));

// ---------------------------------------------------------------------------
// Batched GEMM: C[b,h] = alpha * A[b,h] x B[b,h] (+bias) (+relu)
// A: [M,K] row-major (lda), B: [K,N] (ldb) or [N,K] if TRANS_B.
// Tile: 128x128 per block (256 threads / 8 waves), K-step 32.
// fp32 operands converted ONCE to bf16 at LDS-staging time; fragments are
// pure ds_load_b128 of packed bf16; f32 accumulation in WMMA.
// Register double-buffer prefetches the next K-tile during WMMA.
// ---------------------------------------------------------------------------
template <bool TRANS_B, int EPI>  // EPI: 0=none, 1=+bias, 2=+bias+relu
__global__ __launch_bounds__(256) void gemm_wmma_bf16(
    const float* __restrict__ A, int lda, long sAb, long sAh,
    const float* __restrict__ Bm, int ldb, long sBb, long sBh,
    float* __restrict__ C, int ldc, long sCb, long sCh,
    const float* __restrict__ bias, int M, int N, int K, float alpha, int Hb) {
  __shared__ bf16_t As[128][40];  // [m][k] bf16, row = 80B (16B aligned)
  __shared__ bf16_t Bs[128][40];  // [n][k] bf16

  const int z = blockIdx.z;
  const int bb = z / Hb, hh = z % Hb;
  A  += (long)bb * sAb + (long)hh * sAh;
  Bm += (long)bb * sBb + (long)hh * sBh;
  C  += (long)bb * sCb + (long)hh * sCh;

  const int tid   = threadIdx.x;
  const int lane  = tid & 31;
  const int wid   = tid >> 5;
  const int waveM = wid >> 1;  // 0..3  (32 rows each)
  const int waveN = wid & 1;   // 0..1  (64 cols each)
  const int m0 = blockIdx.y * 128;
  const int n0 = blockIdx.x * 128;
  const int lm = lane & 15;
  const int hi = lane >> 4;    // half-wave select

  v8f acc[2][4] = {};

  float4 ra[4], rb[4];

  auto loadA = [&](int k0) {
#pragma unroll
    for (int i = 0; i < 4; ++i) {
      const int idx = tid + i * 256;  // 0..1023
      const int r = idx >> 3;         // 0..127
      const int c = (idx & 7) << 2;   // 0..28
      float4 v; v.x = v.y = v.z = v.w = 0.f;
      const int gm = m0 + r;
      if (gm < M) v = *(const float4*)(A + (long)gm * lda + k0 + c);
      ra[i] = v;
    }
  };
  auto loadB = [&](int k0) {
#pragma unroll
    for (int i = 0; i < 4; ++i) {
      const int idx = tid + i * 256;
      float4 v; v.x = v.y = v.z = v.w = 0.f;
      if (TRANS_B) {
        const int r = idx >> 3;
        const int c = (idx & 7) << 2;
        const int gn = n0 + r;
        if (gn < N) v = *(const float4*)(Bm + (long)gn * ldb + k0 + c);
      } else {
        const int kr = idx >> 5;        // 0..31
        const int c = (idx & 31) << 2;  // 0..124
        if (n0 + c < N) v = *(const float4*)(Bm + (long)(k0 + kr) * ldb + n0 + c);
      }
      rb[i] = v;
    }
  };
  auto storeTiles = [&]() {
#pragma unroll
    for (int i = 0; i < 4; ++i) {
      const int idx = tid + i * 256;
      {
        const int r = idx >> 3;
        const int c = (idx & 7) << 2;
        v4bf t;
        t[0] = (bf16_t)ra[i].x; t[1] = (bf16_t)ra[i].y;
        t[2] = (bf16_t)ra[i].z; t[3] = (bf16_t)ra[i].w;
        *(v4bf*)&As[r][c] = t;
      }
      if (TRANS_B) {
        const int r = idx >> 3;
        const int c = (idx & 7) << 2;
        v4bf t;
        t[0] = (bf16_t)rb[i].x; t[1] = (bf16_t)rb[i].y;
        t[2] = (bf16_t)rb[i].z; t[3] = (bf16_t)rb[i].w;
        *(v4bf*)&Bs[r][c] = t;
      } else {
        const int kr = idx >> 5;
        const int c = (idx & 31) << 2;
        Bs[c + 0][kr] = (bf16_t)rb[i].x;
        Bs[c + 1][kr] = (bf16_t)rb[i].y;
        Bs[c + 2][kr] = (bf16_t)rb[i].z;
        Bs[c + 3][kr] = (bf16_t)rb[i].w;
      }
    }
  };

  loadA(0);
  loadB(0);

  for (int k0 = 0; k0 < K; k0 += 32) {
    storeTiles();
    __syncthreads();
    // prefetch next K-tile into registers while WMMAs run
    if (k0 + 32 < K) {
      loadA(k0 + 32);
      loadB(k0 + 32);
    }

    // ---- fragment gather (ISA 7.12.2 16-bit layouts), packed bf16 ----
    v16bf af[2];
#pragma unroll
    for (int mi = 0; mi < 2; ++mi) {
      const int row = waveM * 32 + mi * 16 + lm;
      const v8bf lo = *(const v8bf*)&As[row][hi * 8];        // K = kh..kh+7
      const v8bf hs = *(const v8bf*)&As[row][hi * 8 + 16];   // K = kh+16..kh+23
      af[mi] = __builtin_shufflevector(lo, hs, 0, 1, 2, 3, 4, 5, 6, 7, 8, 9,
                                       10, 11, 12, 13, 14, 15);
    }
    v16bf bfv[4];
#pragma unroll
    for (int ni = 0; ni < 4; ++ni) {
      const int col = waveN * 64 + ni * 16 + lm;
      const v8bf lo = *(const v8bf*)&Bs[col][hi * 8];        // K = kh..kh+7
      const v8bf hs = *(const v8bf*)&Bs[col][hi * 8 + 16];   // K = kh+16..kh+23
      bfv[ni] = __builtin_shufflevector(lo, hs, 0, 1, 2, 3, 4, 5, 6, 7, 8, 9,
                                        10, 11, 12, 13, 14, 15);
    }
#pragma unroll
    for (int mi = 0; mi < 2; ++mi)
#pragma unroll
      for (int ni = 0; ni < 4; ++ni)
        acc[mi][ni] = __builtin_amdgcn_wmma_f32_16x16x32_bf16(
            false, af[mi], false, bfv[ni], (short)0, acc[mi][ni], false, false);
    __syncthreads();
  }

  // ---- epilogue ----
#pragma unroll
  for (int mi = 0; mi < 2; ++mi)
#pragma unroll
    for (int ni = 0; ni < 4; ++ni)
#pragma unroll
      for (int r = 0; r < 8; ++r) {
        const int gm = m0 + waveM * 32 + mi * 16 + hi * 8 + r;
        const int gn = n0 + waveN * 64 + ni * 16 + lm;
        if (gm < M && gn < N) {
          float v = acc[mi][ni][r] * alpha;
          if (EPI >= 1) v += bias[gn];
          if (EPI == 2) v = fmaxf(v, 0.f);
          C[(long)gm * ldc + gn] = v;
        }
      }
}

// ---------------------------------------------------------------------------
// Masked softmax over last dim; one wave32 per row. sc: [Bt*H, Lq, Lk]
// mask: int [B, maskLen] per key position; causal adds k<=q.
// ---------------------------------------------------------------------------
__global__ __launch_bounds__(256) void softmax_mask_kernel(
    float* __restrict__ sc, const int* __restrict__ mask, int Lq, int Lk,
    int H, int maskLen, int causal) {
  const int lane = threadIdx.x & 31;
  const int wid = threadIdx.x >> 5;
  const long row = (long)blockIdx.x * 8 + wid;
  const long bh = row / Lq;
  const int q = (int)(row % Lq);
  const int b = (int)(bh / H);
  float* r = sc + row * (long)Lk;
  const int* m = mask + (long)b * maskLen;
  const int nIt = Lk >> 5;  // <= 32
  float vals[32];
  float mx = -3.0e38f;
  for (int i = 0; i < nIt; ++i) {
    const int k = lane + (i << 5);
    const bool ok = (m[k] != 0) && (!causal || (k <= q));
    const float v = ok ? r[k] : -1.0e18f;
    vals[i] = v;
    mx = fmaxf(mx, v);
  }
  for (int o = 16; o; o >>= 1) mx = fmaxf(mx, __shfl_xor(mx, o, 32));
  float s = 0.f;
  for (int i = 0; i < nIt; ++i) {
    vals[i] = __expf(vals[i] - mx);
    s += vals[i];
  }
  for (int o = 16; o; o >>= 1) s += __shfl_xor(s, o, 32);
  const float inv = 1.f / s;
  for (int i = 0; i < nIt; ++i) r[lane + (i << 5)] = vals[i] * inv;
}

// ---------------------------------------------------------------------------
// out = LayerNorm(x + a) * g + b ; one wave32 per row of D=512
// ---------------------------------------------------------------------------
__global__ __launch_bounds__(256) void ln_residual_kernel(
    const float* __restrict__ x, const float* __restrict__ a,
    const float* __restrict__ g, const float* __restrict__ b,
    float* __restrict__ out, int D) {
  const int lane = threadIdx.x & 31;
  const int wid = threadIdx.x >> 5;
  const long row = (long)blockIdx.x * 8 + wid;
  const int nIt = D >> 5;  // 16
  float vals[16];
  float s1 = 0.f, s2 = 0.f;
  for (int i = 0; i < nIt; ++i) {
    const int d = lane + (i << 5);
    const float v = x[row * D + d] + a[row * D + d];
    vals[i] = v;
    s1 += v;
    s2 += v * v;
  }
  for (int o = 16; o; o >>= 1) {
    s1 += __shfl_xor(s1, o, 32);
    s2 += __shfl_xor(s2, o, 32);
  }
  const float mu = s1 / D;
  const float var = s2 / D - mu * mu;
  const float rs = rsqrtf(var + 1e-5f);
  for (int i = 0; i < nIt; ++i) {
    const int d = lane + (i << 5);
    out[row * D + d] = (vals[i] - mu) * rs * g[d] + b[d];
  }
}

// ---------------------------------------------------------------------------
// out[b,l,d] = emb[tok[b,l], d] * scale + PE(l, d)
// ---------------------------------------------------------------------------
__global__ __launch_bounds__(256) void embed_pe_kernel(
    const int* __restrict__ tok, const float* __restrict__ emb,
    float* __restrict__ out, int L, int D, float scale) {
  const long i = (long)blockIdx.x * blockDim.x + threadIdx.x;
  const int d = (int)(i % D);
  const long bl = i / D;
  const int pos = (int)(bl % L);
  const int t = tok[bl];
  const float div = __expf(-(float)(d & ~1) * (9.210340371976184f / (float)D));
  const float ang = (float)pos * div;
  const float pe = (d & 1) ? __cosf(ang) : __sinf(ang);
  out[i] = emb[(long)t * D + d] * scale + pe;
}

// ---------------------------------------------------------------------------
// Host-side GEMM dispatch
// ---------------------------------------------------------------------------
static void gemm(hipStream_t st, const float* A, int lda, long sAb, long sAh,
                 const float* Bm, int ldb, long sBb, long sBh, float* C,
                 int ldc, long sCb, long sCh, const float* bias, int M, int N,
                 int K, float alpha, int batch, int Hb, bool transB, int epi) {
  dim3 g((unsigned)((N + 127) / 128), (unsigned)((M + 127) / 128),
         (unsigned)batch);
#define GEMM_ARGS A, lda, sAb, sAh, Bm, ldb, sBb, sBh, C, ldc, sCb, sCh, bias, M, N, K, alpha, Hb
  if (transB) {
    if (epi == 0)      gemm_wmma_bf16<true, 0><<<g, 256, 0, st>>>(GEMM_ARGS);
    else if (epi == 1) gemm_wmma_bf16<true, 1><<<g, 256, 0, st>>>(GEMM_ARGS);
    else               gemm_wmma_bf16<true, 2><<<g, 256, 0, st>>>(GEMM_ARGS);
  } else {
    if (epi == 0)      gemm_wmma_bf16<false, 0><<<g, 256, 0, st>>>(GEMM_ARGS);
    else if (epi == 1) gemm_wmma_bf16<false, 1><<<g, 256, 0, st>>>(GEMM_ARGS);
    else               gemm_wmma_bf16<false, 2><<<g, 256, 0, st>>>(GEMM_ARGS);
  }
#undef GEMM_ARGS
}

extern "C" void kernel_launch(void* const* d_in, const int* in_sizes, int n_in,
                              void* d_out, int out_size, void* d_ws,
                              size_t ws_size, hipStream_t stream) {
  (void)in_sizes; (void)n_in; (void)out_size; (void)ws_size;
  constexpr int Bv = 4, S = 1024, T = 512, D = 512, H = 8, HD = 64, F = 2048,
                NL = 6, V = 32000;
  const float emb_scale = 22.62741699796952f;     // sqrt(512)
  const float inv_sqrt_d = 0.04419417382415922f;  // 1/sqrt(512)

  const int* src = (const int*)d_in[0];
  const int* src_mask = (const int*)d_in[1];
  const int* trg = (const int*)d_in[2];
  const int* trg_mask = (const int*)d_in[3];
  const float* src_emb = (const float*)d_in[4];
  const float* trg_emb = (const float*)d_in[5];
  // encoder params (dict insertion order), each with leading N_LAYERS dim
  const float* e_wq = (const float*)d_in[6];
  const float* e_bq = (const float*)d_in[7];
  const float* e_wk = (const float*)d_in[8];
  const float* e_bk = (const float*)d_in[9];
  const float* e_wv = (const float*)d_in[10];
  const float* e_bv = (const float*)d_in[11];
  const float* e_wo = (const float*)d_in[12];
  const float* e_bo = (const float*)d_in[13];
  const float* e_w1 = (const float*)d_in[14];
  const float* e_b1 = (const float*)d_in[15];
  const float* e_w2 = (const float*)d_in[16];
  const float* e_b2 = (const float*)d_in[17];
  const float* e_l1g = (const float*)d_in[18];
  const float* e_l1b = (const float*)d_in[19];
  const float* e_l2g = (const float*)d_in[20];
  const float* e_l2b = (const float*)d_in[21];
  // decoder params
  const float* d_sa_wq = (const float*)d_in[22];
  const float* d_sa_bq = (const float*)d_in[23];
  const float* d_sa_wk = (const float*)d_in[24];
  const float* d_sa_bk = (const float*)d_in[25];
  const float* d_sa_wv = (const float*)d_in[26];
  const float* d_sa_bv = (const float*)d_in[27];
  const float* d_sa_wo = (const float*)d_in[28];
  const float* d_sa_bo = (const float*)d_in[29];
  const float* d_ca_wq = (const float*)d_in[30];
  const float* d_ca_bq = (const float*)d_in[31];
  const float* d_ca_wk = (const float*)d_in[32];
  const float* d_ca_bk = (const float*)d_in[33];
  const float* d_ca_wv = (const float*)d_in[34];
  const float* d_ca_bv = (const float*)d_in[35];
  const float* d_ca_wo = (const float*)d_in[36];
  const float* d_ca_bo = (const float*)d_in[37];
  const float* d_w1 = (const float*)d_in[38];
  const float* d_b1 = (const float*)d_in[39];
  const float* d_w2 = (const float*)d_in[40];
  const float* d_b2 = (const float*)d_in[41];
  const float* d_l1g = (const float*)d_in[42];
  const float* d_l1b = (const float*)d_in[43];
  const float* d_l2g = (const float*)d_in[44];
  const float* d_l2b = (const float*)d_in[45];
  const float* d_l3g = (const float*)d_in[46];
  const float* d_l3b = (const float*)d_in[47];
  const float* fc_w = (const float*)d_in[48];
  const float* fc_b = (const float*)d_in[49];

  // workspace carve-up
  float* ws = (float*)d_ws;
  size_t off = 0;
  auto alloc = [&](size_t n) { float* p = ws + off; off += n; return p; };
  float* xe = alloc((size_t)Bv * S * D);      // encoder activations
  float* yd = alloc((size_t)Bv * T * D);      // decoder activations
  float* qb = alloc((size_t)Bv * S * D);
  float* kb = alloc((size_t)Bv * S * D);
  float* vb = alloc((size_t)Bv * S * D);
  float* t1 = alloc((size_t)Bv * S * D);
  float* hf = alloc((size_t)Bv * S * F);      // FFN hidden
  float* sc = alloc((size_t)Bv * H * S * S);  // attention scores

  float* logits = (float*)d_out;
  float* attn_out = logits + (size_t)Bv * T * V;  // final-layer cross attn

  // embeddings + positional encoding
  embed_pe_kernel<<<(Bv * S * D) / 256, 256, 0, stream>>>(src, src_emb, xe, S,
                                                          D, emb_scale);
  embed_pe_kernel<<<(Bv * T * D) / 256, 256, 0, stream>>>(trg, trg_emb, yd, T,
                                                          D, emb_scale);

  // ---------------- encoder ----------------
  for (int l = 0; l < NL; ++l) {
    const long oW = (long)l * D * D, oB = (long)l * D;
    gemm(stream, xe, D, 0, 0, e_wq + oW, D, 0, 0, qb, D, 0, 0, e_bq + oB,
         Bv * S, D, D, 1.f, 1, 1, false, 1);
    gemm(stream, xe, D, 0, 0, e_wk + oW, D, 0, 0, kb, D, 0, 0, e_bk + oB,
         Bv * S, D, D, 1.f, 1, 1, false, 1);
    gemm(stream, xe, D, 0, 0, e_wv + oW, D, 0, 0, vb, D, 0, 0, e_bv + oB,
         Bv * S, D, D, 1.f, 1, 1, false, 1);
    // scores = Q K^T / sqrt(D)   (batched over b,h)
    gemm(stream, qb, D, (long)S * D, HD, kb, D, (long)S * D, HD, sc, S,
         (long)H * S * S, (long)S * S, nullptr, S, S, HD, inv_sqrt_d, Bv * H,
         H, true, 0);
    softmax_mask_kernel<<<(Bv * H * S) / 8, 256, 0, stream>>>(sc, src_mask, S,
                                                              S, H, S, 0);
    // ctx = attn V (writes merged heads via ldc=D)
    gemm(stream, sc, S, (long)H * S * S, (long)S * S, vb, D, (long)S * D, HD,
         t1, D, (long)S * D, HD, nullptr, S, HD, S, 1.f, Bv * H, H, false, 0);
    gemm(stream, t1, D, 0, 0, e_wo + oW, D, 0, 0, qb, D, 0, 0, e_bo + oB,
         Bv * S, D, D, 1.f, 1, 1, false, 1);
    ln_residual_kernel<<<(Bv * S) / 8, 256, 0, stream>>>(xe, qb, e_l1g + oB,
                                                         e_l1b + oB, xe, D);
    // FFN
    gemm(stream, xe, D, 0, 0, e_w1 + (long)l * D * F, F, 0, 0, hf, F, 0, 0,
         e_b1 + (long)l * F, Bv * S, F, D, 1.f, 1, 1, false, 2);
    gemm(stream, hf, F, 0, 0, e_w2 + (long)l * F * D, D, 0, 0, t1, D, 0, 0,
         e_b2 + oB, Bv * S, D, F, 1.f, 1, 1, false, 1);
    ln_residual_kernel<<<(Bv * S) / 8, 256, 0, stream>>>(xe, t1, e_l2g + oB,
                                                         e_l2b + oB, xe, D);
  }

  // ---------------- decoder ----------------
  for (int l = 0; l < NL; ++l) {
    const long oW = (long)l * D * D, oB = (long)l * D;
    // --- masked self-attention on yd ---
    gemm(stream, yd, D, 0, 0, d_sa_wq + oW, D, 0, 0, qb, D, 0, 0, d_sa_bq + oB,
         Bv * T, D, D, 1.f, 1, 1, false, 1);
    gemm(stream, yd, D, 0, 0, d_sa_wk + oW, D, 0, 0, kb, D, 0, 0, d_sa_bk + oB,
         Bv * T, D, D, 1.f, 1, 1, false, 1);
    gemm(stream, yd, D, 0, 0, d_sa_wv + oW, D, 0, 0, vb, D, 0, 0, d_sa_bv + oB,
         Bv * T, D, D, 1.f, 1, 1, false, 1);
    gemm(stream, qb, D, (long)T * D, HD, kb, D, (long)T * D, HD, sc, T,
         (long)H * T * T, (long)T * T, nullptr, T, T, HD, inv_sqrt_d, Bv * H,
         H, true, 0);
    softmax_mask_kernel<<<(Bv * H * T) / 8, 256, 0, stream>>>(sc, trg_mask, T,
                                                              T, H, T, 1);
    gemm(stream, sc, T, (long)H * T * T, (long)T * T, vb, D, (long)T * D, HD,
         t1, D, (long)T * D, HD, nullptr, T, HD, T, 1.f, Bv * H, H, false, 0);
    gemm(stream, t1, D, 0, 0, d_sa_wo + oW, D, 0, 0, qb, D, 0, 0,
         d_sa_bo + oB, Bv * T, D, D, 1.f, 1, 1, false, 1);
    ln_residual_kernel<<<(Bv * T) / 8, 256, 0, stream>>>(yd, qb, d_l1g + oB,
                                                         d_l1b + oB, yd, D);
    // --- cross-attention: Q from yd, K/V from encoder output xe ---
    gemm(stream, yd, D, 0, 0, d_ca_wq + oW, D, 0, 0, qb, D, 0, 0, d_ca_bq + oB,
         Bv * T, D, D, 1.f, 1, 1, false, 1);
    gemm(stream, xe, D, 0, 0, d_ca_wk + oW, D, 0, 0, kb, D, 0, 0, d_ca_bk + oB,
         Bv * S, D, D, 1.f, 1, 1, false, 1);
    gemm(stream, xe, D, 0, 0, d_ca_wv + oW, D, 0, 0, vb, D, 0, 0, d_ca_bv + oB,
         Bv * S, D, D, 1.f, 1, 1, false, 1);
    float* cs = (l == NL - 1) ? attn_out : sc;  // last layer attn -> d_out
    gemm(stream, qb, D, (long)T * D, HD, kb, D, (long)S * D, HD, cs, S,
         (long)H * T * S, (long)T * S, nullptr, T, S, HD, inv_sqrt_d, Bv * H,
         H, true, 0);
    softmax_mask_kernel<<<(Bv * H * T) / 8, 256, 0, stream>>>(cs, src_mask, T,
                                                              S, H, S, 0);
    gemm(stream, cs, S, (long)H * T * S, (long)T * S, vb, D, (long)S * D, HD,
         t1, D, (long)T * D, HD, nullptr, T, HD, S, 1.f, Bv * H, H, false, 0);
    gemm(stream, t1, D, 0, 0, d_ca_wo + oW, D, 0, 0, qb, D, 0, 0,
         d_ca_bo + oB, Bv * T, D, D, 1.f, 1, 1, false, 1);
    ln_residual_kernel<<<(Bv * T) / 8, 256, 0, stream>>>(yd, qb, d_l2g + oB,
                                                         d_l2b + oB, yd, D);
    // --- FFN ---
    gemm(stream, yd, D, 0, 0, d_w1 + (long)l * D * F, F, 0, 0, hf, F, 0, 0,
         d_b1 + (long)l * F, Bv * T, F, D, 1.f, 1, 1, false, 2);
    gemm(stream, hf, F, 0, 0, d_w2 + (long)l * F * D, D, 0, 0, t1, D, 0, 0,
         d_b2 + oB, Bv * T, D, F, 1.f, 1, 1, false, 1);
    ln_residual_kernel<<<(Bv * T) / 8, 256, 0, stream>>>(yd, t1, d_l3g + oB,
                                                         d_l3b + oB, yd, D);
  }

  // final vocab projection -> logits in d_out
  gemm(stream, yd, D, 0, 0, fc_w, V, 0, 0, logits, V, 0, 0, fc_b, Bv * T, V, D,
       1.f, 1, 1, false, 1);
}
